// SiglipAttention_7215545057937
// MI455X (gfx1250) — matrix operand order repaired and checked by reading, not compiled, not run on hardware
//
#include <hip/hip_runtime.h>
#include <hip/hip_bf16.h>

typedef __bf16 bf16;
typedef __attribute__((ext_vector_type(8)))  bf16  v8bf;
typedef __attribute__((ext_vector_type(16))) bf16  v16bf;
typedef __attribute__((ext_vector_type(8)))  float v8f;

#define EMBED 1152
#define HEADS 16
#define HDIM  72
#define HPAD  96          // head dim padded to multiple of 32 (zero pad)
#define SEQ   729
#define SEQP  736         // key rows padded to multiple of 32
#define BATCH 16
#define MTOT  (BATCH * SEQ)   // 11664
#define NROWTILES 46          // ceil(729/16) query-row tiles in attention
#define NCOLTILES 46          // ceil(736/16) key-column tiles

// ---------------------------------------------------------------------------
// WMMA helpers (CDNA5, wave32, 16x16x32 bf16 -> f32)
// ---------------------------------------------------------------------------
__device__ __forceinline__ v8f wmma_bf16(v16bf a, v16bf b, v8f c) {
  // (neg_a, A, neg_b, B, c_mod, C, reuse_a, reuse_b)
  return __builtin_amdgcn_wmma_f32_16x16x32_bf16(false, a, false, b,
                                                 (short)0, c, false, false);
}

// A-matrix 16x32 bf16 fragment (ISA 7.12.2):
// lane: row M = lane&15; K chunk base kh = 8*(lane>>4); elems 0..7 at K=kh..kh+7,
// elems 8..15 at K=kh+16..kh+23. Row-major source with leading dim `ld`.
__device__ __forceinline__ v16bf ldfragA(const bf16* __restrict__ base, int ld,
                                         int row0, int kb, int lane, int rowmax) {
  int r = row0 + (lane & 15);
  r = (r > rowmax) ? rowmax : r;          // clamp (stores are guarded)
  int k0 = kb + ((lane >> 4) << 3);
  const bf16* p = base + (long)r * ld + k0;
  v8bf lo = *(const v8bf*)(p);
  v8bf hi = *(const v8bf*)(p + 16);
  return __builtin_shufflevector(lo, hi, 0, 1, 2, 3, 4, 5, 6, 7,
                                         8, 9, 10, 11, 12, 13, 14, 15);
}

// B-matrix 32x16 (KxN) bf16 fragment, loaded column-major (ISA 7.12.3):
// lane: col N = lane&15; K chunk base kh = 8*(lane>>4); elems 0..7 at K=kh..kh+7,
// elems 8..15 at K=kh+16..kh+23 (same order as A).
// Source stored as rows of length `ld` per N (i.e. W[n,k] / K-matrix / V^T).
__device__ __forceinline__ v16bf ldfragB(const bf16* __restrict__ base, int ld,
                                         int col0, int kb, int lane, int colmax) {
  int n = col0 + (lane & 15);
  n = (n > colmax) ? colmax : n;
  int k0 = kb + ((lane >> 4) << 3);
  const bf16* pb = base + (long)n * ld + k0;
  v8bf lo = *(const v8bf*)(pb);
  v8bf hi = *(const v8bf*)(pb + 16);
  return __builtin_shufflevector(lo, hi, 0, 1, 2, 3, 4, 5, 6, 7,
                                         8, 9, 10, 11, 12, 13, 14, 15);
}

// ---------------------------------------------------------------------------
// f32 -> bf16 staging
// ---------------------------------------------------------------------------
__global__ void cvt_kernel(const float* __restrict__ in, bf16* __restrict__ out,
                           long n) {
  long i = (long)blockIdx.x * 256 + threadIdx.x;
  if (i < n) out[i] = (bf16)in[i];
}

// ---------------------------------------------------------------------------
// Generic bf16 WMMA GEMM:  C[m,n] = sum_k A[m,k]*W[n,k] + bias[n]
// block: 256 threads = 8 waves (4 x 2), block tile 128x128, wave tile 32x64.
// MODE 0: store f32 row-major [M, EMBED]          (final projection)
// MODE 1: store bf16 Q layout  [b,h,n,HPAD]
// MODE 2: store bf16 K layout  [b,h,n,HPAD]
// MODE 3: store bf16 V^T layout[b,h,d(HPAD),SEQP]
// ---------------------------------------------------------------------------
template <int MODE>
__global__ __launch_bounds__(256) void gemm_kernel(
    const bf16* __restrict__ A, const bf16* __restrict__ Bw,
    const float* __restrict__ bias, int M,
    float* __restrict__ outF, bf16* __restrict__ outB) {
  const int K = EMBED;
  int lane = threadIdx.x & 31;
  int wave = threadIdx.x >> 5;
  int wm = wave & 3, wn = wave >> 2;
  int tm = blockIdx.y * 128 + wm * 32;
  int tn = blockIdx.x * 128 + wn * 64;

  v8f acc[2][4];
#pragma unroll
  for (int i = 0; i < 2; i++)
#pragma unroll
    for (int j = 0; j < 4; j++)
#pragma unroll
      for (int e = 0; e < 8; e++) acc[i][j][e] = 0.f;

  for (int kb = 0; kb < K; kb += 32) {
    if (kb + 32 < K) {   // speculative prefetch of next K-slab (global_prefetch)
      __builtin_prefetch(A + (long)tm * K + kb + 32, 0, 1);
      __builtin_prefetch(Bw + (long)tn * K + kb + 32, 0, 1);
    }
    v16bf a0 = ldfragA(A, K, tm, kb, lane, M - 1);
    v16bf a1 = ldfragA(A, K, tm + 16, kb, lane, M - 1);
#pragma unroll
    for (int j = 0; j < 4; j++) {
      v16bf bf = ldfragB(Bw, K, tn + j * 16, kb, lane, EMBED - 1);
      acc[0][j] = wmma_bf16(a0, bf, acc[0][j]);
      acc[1][j] = wmma_bf16(a1, bf, acc[1][j]);
    }
  }

  int cl = lane & 15;
  int rb = (lane >> 4) << 3;
#pragma unroll
  for (int i = 0; i < 2; i++) {
#pragma unroll
    for (int j = 0; j < 4; j++) {
      int n = tn + j * 16 + cl;
      float bb = bias[n];
#pragma unroll
      for (int r = 0; r < 8; r++) {
        int m = tm + i * 16 + rb + r;
        if (m >= M) continue;
        float v = acc[i][j][r] + bb;
        if constexpr (MODE == 0) {
          outF[(long)m * EMBED + n] = v;
        } else {
          int bidx = m / SEQ, nn = m - bidx * SEQ;
          int hh = n / HDIM, dd = n - hh * HDIM;
          if constexpr (MODE == 1 || MODE == 2) {
            outB[(((long)(bidx * HEADS + hh)) * SEQ + nn) * HPAD + dd] = (bf16)v;
          } else {
            outB[(((long)(bidx * HEADS + hh)) * HPAD + dd) * SEQP + nn] = (bf16)v;
          }
        }
      }
    }
  }
}

// ---------------------------------------------------------------------------
// Attention core: per (b,h, 16-row query tile)
//   S = Q K^T / sqrt(D) -> LDS, row softmax -> attn (f32 out), ctx = P V -> bf16
// All acc[] indexing is compile-time constant (fully unrolled, wave-uniform
// guards) so accumulators live in VGPRs -- no scratch spills.
// ---------------------------------------------------------------------------
__global__ __launch_bounds__(256) void attn_kernel(
    const bf16* __restrict__ Q, const bf16* __restrict__ Kst,
    const bf16* __restrict__ Vt, float* __restrict__ attnOut,
    bf16* __restrict__ ctxOut) {
  __shared__ float sS[16][744];   // 16 rows x 736 cols (+8 pad) f32 = 46.5 KB

  int rt = blockIdx.x % NROWTILES;
  int bh = blockIdx.x / NROWTILES;
  int h = bh & 15;
  int b = bh >> 4;
  const bf16* q  = Q   + (long)bh * SEQ * HPAD;
  const bf16* kk = Kst + (long)bh * SEQ * HPAD;
  const bf16* vt = Vt  + (long)bh * HPAD * SEQP;
  int row0 = rt * 16;
  int lane = threadIdx.x & 31;
  int wave = threadIdx.x >> 5;
  const float scale = 0.11785113019775793f;   // 1/sqrt(72)

  // ---- phase 1: S tile = Q(16 x 96) * K^T(96 x 736), 46 col tiles / 8 waves
  v8f acc[6];
#pragma unroll
  for (int t = 0; t < 6; t++)
#pragma unroll
    for (int e = 0; e < 8; e++) acc[t][e] = 0.f;

#pragma unroll
  for (int kb = 0; kb < HPAD; kb += 32) {
    v16bf a0 = ldfragA(q, HPAD, row0, kb, lane, SEQ - 1);
#pragma unroll
    for (int t = 0; t < 6; t++) {
      int ct = wave + t * 8;                 // wave-uniform guard: EXEC stays
      if (ct < NCOLTILES) {                  // all-ones inside the wave
        v16bf bf = ldfragB(kk, HPAD, ct * 16, kb, lane, SEQ - 1);
        acc[t] = wmma_bf16(a0, bf, acc[t]);
      }
    }
  }
  int cl = lane & 15;
  int rb = (lane >> 4) << 3;
#pragma unroll
  for (int t = 0; t < 6; t++) {
    int ct = wave + t * 8;
    if (ct < NCOLTILES) {
      int c = ct * 16 + cl;
#pragma unroll
      for (int r = 0; r < 8; r++) sS[rb + r][c] = acc[t][r] * scale;
    }
  }
  __syncthreads();

  // ---- phase 2: softmax per row (16 threads per row), write attn f32
  {
    int row = threadIdx.x >> 4;
    int l = threadIdx.x & 15;
    float mx = -1e30f;
    for (int c = l; c < SEQ; c += 16) mx = fmaxf(mx, sS[row][c]);
#pragma unroll
    for (int m = 1; m < 16; m <<= 1) mx = fmaxf(mx, __shfl_xor(mx, m, 16));
    float sum = 0.f;
    for (int c = l; c < SEQP; c += 16) {
      float pv = (c < SEQ) ? __expf(sS[row][c] - mx) : 0.f;  // mask pad cols
      sS[row][c] = pv;
      sum += pv;
    }
#pragma unroll
    for (int m = 1; m < 16; m <<= 1) sum += __shfl_xor(sum, m, 16);
    float inv = 1.f / sum;
    int rg = row0 + row;
    bool valid = (rg < SEQ);
    float* ao = attnOut + ((long)bh * SEQ + rg) * (long)SEQ;
    for (int c = l; c < SEQ; c += 16) {
      float pv = sS[row][c] * inv;
      sS[row][c] = pv;
      if (valid) ao[c] = pv;
    }
  }
  __syncthreads();

  // ---- phase 3: ctx(16 x 72) = P(16 x 736) * V(736 x 72); 5 col tiles of 16
  if (wave < 5) {                            // wave-uniform
    int ct = wave;
    v8f cacc;
#pragma unroll
    for (int e = 0; e < 8; e++) cacc[e] = 0.f;
    for (int j = 0; j < SEQP; j += 32) {
      // A fragment of P straight from LDS, converted to bf16
      int ar = lane & 15;
      int kh = (lane >> 4) << 3;
      v8f f0 = *(const v8f*)&sS[ar][j + kh];
      v8f f1 = *(const v8f*)&sS[ar][j + kh + 16];
      v16bf af;
#pragma unroll
      for (int e = 0; e < 8; e++) {
        af[e] = (bf16)f0[e];
        af[e + 8] = (bf16)f1[e];
      }
      v16bf bfb = ldfragB(vt, SEQP, ct * 16, j, lane, HPAD - 1);
      cacc = wmma_bf16(af, bfb, cacc);
    }
    int d = ct * 16 + cl;
    if (d < HDIM) {
#pragma unroll
      for (int r = 0; r < 8; r++) {
        int rg = row0 + rb + r;
        if (rg < SEQ)
          ctxOut[((long)(b * SEQ + rg)) * EMBED + h * HDIM + d] = (bf16)cacc[r];
      }
    }
  }
}

// ---------------------------------------------------------------------------
extern "C" void kernel_launch(void* const* d_in, const int* in_sizes, int n_in,
                              void* d_out, int out_size, void* d_ws, size_t ws_size,
                              hipStream_t stream) {
  (void)in_sizes; (void)n_in; (void)out_size; (void)ws_size;
  const float* x  = (const float*)d_in[0];
  const float* Wq = (const float*)d_in[1];
  const float* bq = (const float*)d_in[2];
  const float* Wk = (const float*)d_in[3];
  const float* bk = (const float*)d_in[4];
  const float* Wv = (const float*)d_in[5];
  const float* bv = (const float*)d_in[6];
  const float* Wo = (const float*)d_in[7];
  const float* bo = (const float*)d_in[8];

  float* out  = (float*)d_out;                    // [B, N, E] f32
  float* attn = out + (long)MTOT * EMBED;         // [B, H, N, N] f32

  // workspace layout (bf16 staging)
  bf16* p = (bf16*)d_ws;
  bf16* xb   = p; p += (long)MTOT * EMBED;                  // x bf16
  bf16* wqb  = p; p += (long)EMBED * EMBED;
  bf16* wkb  = p; p += (long)EMBED * EMBED;
  bf16* wvb  = p; p += (long)EMBED * EMBED;
  bf16* wob  = p; p += (long)EMBED * EMBED;
  bf16* qs   = p; p += (long)BATCH * HEADS * SEQ * HPAD;    // Q [b,h,n,96]
  bf16* ks   = p; p += (long)BATCH * HEADS * SEQ * HPAD;    // K [b,h,n,96]
  bf16* vts  = p; p += (long)BATCH * HEADS * HPAD * SEQP;   // V^T [b,h,96,736]
  bf16* ctxb = p;                                           // ctx [b,n,1152]

  // zero Q/K/V^T staging so the padded head-dim / key-row lanes contribute 0
  size_t zbytes = ((size_t)2 * BATCH * HEADS * SEQ * HPAD +
                   (size_t)BATCH * HEADS * HPAD * SEQP) * sizeof(bf16);
  hipMemsetAsync(qs, 0, zbytes, stream);

  long nx = (long)MTOT * EMBED;
  long nw = (long)EMBED * EMBED;
  cvt_kernel<<<dim3((unsigned)((nx + 255) / 256)), dim3(256), 0, stream>>>(x, xb, nx);
  cvt_kernel<<<dim3((unsigned)((nw + 255) / 256)), dim3(256), 0, stream>>>(Wq, wqb, nw);
  cvt_kernel<<<dim3((unsigned)((nw + 255) / 256)), dim3(256), 0, stream>>>(Wk, wkb, nw);
  cvt_kernel<<<dim3((unsigned)((nw + 255) / 256)), dim3(256), 0, stream>>>(Wv, wvb, nw);
  cvt_kernel<<<dim3((unsigned)((nw + 255) / 256)), dim3(256), 0, stream>>>(Wo, wob, nw);

  dim3 ggrid(EMBED / 128, (MTOT + 127) / 128);   // (9, 92)
  gemm_kernel<1><<<ggrid, 256, 0, stream>>>(xb, wqb, bq, MTOT, nullptr, qs);
  gemm_kernel<2><<<ggrid, 256, 0, stream>>>(xb, wkb, bk, MTOT, nullptr, ks);
  gemm_kernel<3><<<ggrid, 256, 0, stream>>>(xb, wvb, bv, MTOT, nullptr, vts);

  attn_kernel<<<dim3(BATCH * HEADS * NROWTILES), 256, 0, stream>>>(
      qs, ks, vts, attn, ctxb);

  gemm_kernel<0><<<ggrid, 256, 0, stream>>>(ctxb, wob, bo, MTOT, out, nullptr);
}
